// EGNN_36249523978651
// MI455X (gfx1250) — matrix operand order repaired and checked by reading, not compiled, not run on hardware
//
#include <hip/hip_runtime.h>
#include <cstdint>

typedef __attribute__((ext_vector_type(16))) _Float16 v16h;
typedef __attribute__((ext_vector_type(8)))  float    v8f;

#define NNODE      14848
#define NEDGE      415744
#define HID        128
#define NBATCH     512
#define NPG        29
#define K_EDGE     259
#define K_EDGE_PAD 288
#define K_NODE     271
#define K_NODE_PAD 288
#define EDGE_TILES (NEDGE / 16)   // 25984
#define NODE_TILES (NNODE / 16)   // 928
#define EDGE_GRID  3248           // 8 tiles per block
#define NODE_GRID  464            // 2 tiles per block
// LDS row strides (elements), chosen so row stride in dwords is coprime with 64 banks
#define STRIDE_E   290   // 580B = 145 dwords, gcd(145,64)=1
#define STRIDE_H   130   // 260B = 65 dwords,  gcd(65,64)=1
#define STRIDE_EMB 34    // 68B  = 17 dwords,  gcd(17,64)=1

union AFrag { v16h v; unsigned int u[8]; };

// A fragment (16x32 f16) from LDS row-major tile.
// Layout (CDNA5 ISA, 16-bit A 16x32): lane L holds row m=L%16, half g=L/16;
// element e -> K = (e/8)*16 + g*8 + (e%8). Consecutive e-pairs are K-adjacent -> dword loads.
__device__ __forceinline__ v16h load_a_lds(const _Float16* base, int stride, int m, int g, int kbase) {
  const unsigned int* p = (const unsigned int*)(base + m * stride + kbase + g * 8);
  AFrag f;
  f.u[0] = p[0];  f.u[1] = p[1];  f.u[2] = p[2];  f.u[3] = p[3];
  f.u[4] = p[8];  f.u[5] = p[9];  f.u[6] = p[10]; f.u[7] = p[11];
  return f.v;
}

// B fragment: pre-swizzled by pack_weight so each lane reads 16 contiguous f16 (32B).
__device__ __forceinline__ v16h load_b_packed(const _Float16* wp, int tile, int lane) {
  return *(const v16h*)(wp + ((size_t)(tile * 32 + lane)) * 16);
}

__device__ __forceinline__ v8f wmma16(v16h a, v16h b, v8f c) {
  return __builtin_amdgcn_wmma_f32_16x16x32_f16(false, a, false, b, (short)0, c, false, false);
}

__device__ __forceinline__ float silu_f(float x) { return x / (1.0f + __expf(-x)); }

// ---- weight packing: fp32 (K x 128) -> per-lane WMMA B fragments (f16), K zero-padded ----
// B layout (CDNA5 ISA B 32x16 per 64x16 pattern): lane L holds col n=L%16,
// group g=L/16; element e -> K_local = (e/8)*16 + g*8 + (e%8).
__global__ __launch_bounds__(256) void pack_weight(const float* __restrict__ src,
                                                   _Float16* __restrict__ dst,
                                                   int K, int Kpad) {
  int idx = blockIdx.x * 256 + threadIdx.x;
  if (idx >= Kpad * HID) return;
  int e    = idx & 15;
  int lane = (idx >> 4) & 31;
  int tile = idx >> 9;           // tile = kc*8 + nc
  int kc = tile >> 3, nc = tile & 7;
  int g  = lane >> 4;
  int k  = kc * 32 + (e >> 3) * 16 + g * 8 + (e & 7);
  int n  = nc * 16 + (lane & 15);
  float v = (k < K) ? src[(size_t)k * HID + n] : 0.0f;
  dst[idx] = (_Float16)v;
}

__global__ __launch_bounds__(256) void conv_edges(const long long* __restrict__ e64,
                                                  int* __restrict__ e32, int n) {
  int i = blockIdx.x * 256 + threadIdx.x;
  if (i < n) e32[i] = (int)e64[i];
}

__global__ __launch_bounds__(256) void radial_kernel(const float* __restrict__ x,
                                                     const int* __restrict__ rows,
                                                     const int* __restrict__ cols,
                                                     float* __restrict__ radial) {
  int i = blockIdx.x * 256 + threadIdx.x;
  if (i >= NEDGE) return;
  int r = rows[i], c = cols[i];
  float dx = x[r*3+0] - x[c*3+0];
  float dy = x[r*3+1] - x[c*3+1];
  float dz = x[r*3+2] - x[c*3+2];
  radial[i] = dx*dx + dy*dy + dz*dz;
}

// ---- h = h0 @ Wemb + bemb  (K=15 padded to 32) ----
__global__ __launch_bounds__(256) void embed_kernel(const float* __restrict__ h0,
                                                    const _Float16* __restrict__ Wp,
                                                    const float* __restrict__ b,
                                                    float* __restrict__ hF,
                                                    _Float16* __restrict__ hB) {
  __shared__ _Float16 ldsA[16 * STRIDE_EMB];
  int tid = threadIdx.x, lane = tid & 31, w = tid >> 5;
  int t = blockIdx.x;
  for (int idx = tid; idx < 16 * 32; idx += 256) {
    int i = idx >> 5, k = idx & 31;
    float v = (k < 15) ? h0[(size_t)(t*16 + i) * 15 + k] : 0.0f;
    ldsA[i * STRIDE_EMB + k] = (_Float16)v;
  }
  __syncthreads();
  int nloc = lane & 15, g = lane >> 4, n = w*16 + nloc, m = nloc;
  float bv = b[n];
  v8f acc = {bv, bv, bv, bv, bv, bv, bv, bv};
  v16h bfr = load_b_packed(Wp, w, lane);
  acc = wmma16(load_a_lds(ldsA, STRIDE_EMB, m, g, 0), bfr, acc);
  #pragma unroll
  for (int r = 0; r < 8; ++r) {
    size_t nm = (size_t)(t*16 + r + g*8);
    float vv = acc[r];
    hF[nm * HID + n] = vv;
    hB[nm * HID + n] = (_Float16)vv;
  }
}

// ---- edge MLP (persistent, grid-stride over 16-edge tiles):
//   m = silu(W2 @ silu(W1 @ [h_r|h_c|radial|ea] + b1) + b2) * mask; agg[row] += m ----
__global__ __launch_bounds__(256) void edge_kernel(const _Float16* __restrict__ hB,
                                                   const float* __restrict__ radial,
                                                   const float* __restrict__ edge_attr,
                                                   const float* __restrict__ edge_mask,
                                                   const int* __restrict__ rows,
                                                   const int* __restrict__ cols,
                                                   const _Float16* __restrict__ W1,
                                                   const float* __restrict__ b1,
                                                   const _Float16* __restrict__ W2,
                                                   const float* __restrict__ b2,
                                                   float* __restrict__ agg) {
  __shared__ _Float16 ldsA[16 * STRIDE_E];
  __shared__ _Float16 ldsM[16 * STRIDE_H];
  __shared__ int   ldsRow[16], ldsCol[16];
  __shared__ float ldsMask[16];
  int tid = threadIdx.x, lane = tid & 31, w = tid >> 5;
  int nloc = lane & 15, g = lane >> 4, n = w*16 + nloc, m = nloc;
  int i = tid >> 4, j = tid & 15;

  // Weight fragments + biases live in VGPRs for all tiles handled by this block.
  v16h B1[9], B2[4];
  #pragma unroll
  for (int kc = 0; kc < 9; ++kc) B1[kc] = load_b_packed(W1, kc*8 + w, lane);
  #pragma unroll
  for (int kc = 0; kc < 4; ++kc) B2[kc] = load_b_packed(W2, kc*8 + w, lane);
  float bv1 = b1[n];
  float bv2 = b2[n];

  for (int t = blockIdx.x; t < EDGE_TILES; t += EDGE_GRID) {
    if (tid == 0) {
      int tn = t + EDGE_GRID;                    // prefetch next tile's index data
      if (tn < EDGE_TILES) {
        int egn = tn * 16;
        __builtin_prefetch(&rows[egn], 0, 3);
        __builtin_prefetch(&cols[egn], 0, 3);
        __builtin_prefetch(&edge_mask[egn], 0, 3);
        __builtin_prefetch(&radial[egn], 0, 3);
      }
    }
    if (tid < 16) {
      int eg = t*16 + tid;
      ldsRow[tid]  = rows[eg];
      ldsCol[tid]  = cols[eg];
      ldsMask[tid] = edge_mask[eg];
    }
    __syncthreads();
    {
      const unsigned int* hr = (const unsigned int*)(hB + (size_t)ldsRow[i] * HID);
      const unsigned int* hc = (const unsigned int*)(hB + (size_t)ldsCol[i] * HID);
      unsigned int* dst = (unsigned int*)(ldsA + i * STRIDE_E);
      #pragma unroll
      for (int it = 0; it < 4; ++it) {
        dst[j + 16*it]      = hr[j + 16*it];   // h[row] -> cols 0..127
        dst[64 + j + 16*it] = hc[j + 16*it];   // h[col] -> cols 128..255
      }
      int eg = t*16 + i;
      int k0 = 256 + 2*j;                       // tail cols 256..287
      float v0 = 0.0f, v1 = 0.0f;
      if (k0 == 256)      { v0 = radial[eg];              v1 = edge_attr[(size_t)eg*2 + 0]; }
      else if (k0 == 258) { v0 = edge_attr[(size_t)eg*2 + 1]; }
      _Float16* tb = ldsA + i * STRIDE_E + k0;
      tb[0] = (_Float16)v0; tb[1] = (_Float16)v1;
    }
    __syncthreads();

    v8f acc = {bv1, bv1, bv1, bv1, bv1, bv1, bv1, bv1};
    #pragma unroll
    for (int kc = 0; kc < 9; ++kc)
      acc = wmma16(load_a_lds(ldsA, STRIDE_E, m, g, kc*32), B1[kc], acc);
    #pragma unroll
    for (int r = 0; r < 8; ++r)
      ldsM[(r + g*8) * STRIDE_H + n] = (_Float16)silu_f(acc[r]);
    __syncthreads();

    v8f acc2 = {bv2, bv2, bv2, bv2, bv2, bv2, bv2, bv2};
    #pragma unroll
    for (int kc = 0; kc < 4; ++kc)
      acc2 = wmma16(load_a_lds(ldsM, STRIDE_H, m, g, kc*32), B2[kc], acc2);
    #pragma unroll
    for (int r = 0; r < 8; ++r) {
      int mi = r + g*8;
      float s = silu_f(acc2[r]) * ldsMask[mi];
      atomicAdd(&agg[(size_t)ldsRow[mi] * HID + n], s);   // segment_sum
    }
    __syncthreads();   // protect ldsRow/ldsMask/ldsA before next tile's writes
  }
}

// ---- node MLP (persistent): h += W2 @ silu(W1 @ [h|agg|h0] + b1) + b2 ----
__global__ __launch_bounds__(256) void node_kernel(const float* __restrict__ h0,
                                                   const float* __restrict__ agg,
                                                   const _Float16* __restrict__ W1,
                                                   const float* __restrict__ b1,
                                                   const _Float16* __restrict__ W2,
                                                   const float* __restrict__ b2,
                                                   float* __restrict__ hF,
                                                   _Float16* __restrict__ hB) {
  __shared__ _Float16 ldsA[16 * STRIDE_E];
  __shared__ _Float16 ldsM[16 * STRIDE_H];
  int tid = threadIdx.x, lane = tid & 31, w = tid >> 5;
  int nloc = lane & 15, g = lane >> 4, n = w*16 + nloc, m = nloc;

  v16h B1[9], B2[4];
  #pragma unroll
  for (int kc = 0; kc < 9; ++kc) B1[kc] = load_b_packed(W1, kc*8 + w, lane);
  #pragma unroll
  for (int kc = 0; kc < 4; ++kc) B2[kc] = load_b_packed(W2, kc*8 + w, lane);
  float bv1 = b1[n];
  float bv2 = b2[n];

  for (int t = blockIdx.x; t < NODE_TILES; t += NODE_GRID) {
    for (int idx = tid; idx < 16 * K_NODE_PAD; idx += 256) {
      int i = idx / K_NODE_PAD, k = idx % K_NODE_PAD;
      size_t nm = (size_t)(t*16 + i);
      _Float16 hv;
      if (k < 128)      hv = hB[nm * HID + k];
      else if (k < 256) hv = (_Float16)agg[nm * HID + (k - 128)];
      else if (k < 271) hv = (_Float16)h0[nm * 15 + (k - 256)];
      else              hv = (_Float16)0.0f;
      ldsA[i * STRIDE_E + k] = hv;
    }
    __syncthreads();

    v8f acc = {bv1, bv1, bv1, bv1, bv1, bv1, bv1, bv1};
    #pragma unroll
    for (int kc = 0; kc < 9; ++kc)
      acc = wmma16(load_a_lds(ldsA, STRIDE_E, m, g, kc*32), B1[kc], acc);
    #pragma unroll
    for (int r = 0; r < 8; ++r)
      ldsM[(r + g*8) * STRIDE_H + n] = (_Float16)silu_f(acc[r]);
    __syncthreads();

    v8f acc2 = {bv2, bv2, bv2, bv2, bv2, bv2, bv2, bv2};
    #pragma unroll
    for (int kc = 0; kc < 4; ++kc)
      acc2 = wmma16(load_a_lds(ldsM, STRIDE_H, m, g, kc*32), B2[kc], acc2);
    #pragma unroll
    for (int r = 0; r < 8; ++r) {
      size_t nm = (size_t)(t*16 + r + g*8);
      float nv = hF[nm * HID + n] + acc2[r];     // residual (no outer silu)
      hF[nm * HID + n] = nv;
      hB[nm * HID + n] = (_Float16)nv;
    }
    __syncthreads();
  }
}

// ---- node_dec: t = W2 @ silu(W1 @ h + b1) + b2; hsum[graph] += t * node_mask ----
__global__ __launch_bounds__(256) void node_dec_kernel(const _Float16* __restrict__ hB,
                                                       const float* __restrict__ node_mask,
                                                       const _Float16* __restrict__ W1,
                                                       const float* __restrict__ b1,
                                                       const _Float16* __restrict__ W2,
                                                       const float* __restrict__ b2,
                                                       float* __restrict__ hsum) {
  __shared__ _Float16 ldsA[16 * STRIDE_H];
  __shared__ _Float16 ldsM[16 * STRIDE_H];
  int tid = threadIdx.x, lane = tid & 31, w = tid >> 5;
  int t = blockIdx.x;
  for (int idx = tid; idx < 16 * HID; idx += 256) {
    int i = idx >> 7, k = idx & 127;
    ldsA[i * STRIDE_H + k] = hB[(size_t)(t*16 + i) * HID + k];
  }
  v16h B1[4], B2[4];
  #pragma unroll
  for (int kc = 0; kc < 4; ++kc) B1[kc] = load_b_packed(W1, kc*8 + w, lane);
  #pragma unroll
  for (int kc = 0; kc < 4; ++kc) B2[kc] = load_b_packed(W2, kc*8 + w, lane);
  __syncthreads();

  int nloc = lane & 15, g = lane >> 4, n = w*16 + nloc, m = nloc;
  float bv = b1[n];
  v8f acc = {bv, bv, bv, bv, bv, bv, bv, bv};
  #pragma unroll
  for (int kc = 0; kc < 4; ++kc)
    acc = wmma16(load_a_lds(ldsA, STRIDE_H, m, g, kc*32), B1[kc], acc);
  #pragma unroll
  for (int r = 0; r < 8; ++r)
    ldsM[(r + g*8) * STRIDE_H + n] = (_Float16)silu_f(acc[r]);
  __syncthreads();

  float bv2 = b2[n];
  v8f acc2 = {bv2, bv2, bv2, bv2, bv2, bv2, bv2, bv2};
  #pragma unroll
  for (int kc = 0; kc < 4; ++kc)
    acc2 = wmma16(load_a_lds(ldsM, STRIDE_H, m, g, kc*32), B2[kc], acc2);
  #pragma unroll
  for (int r = 0; r < 8; ++r) {
    int nmi = t*16 + r + g*8;
    float s = acc2[r] * node_mask[nmi];         // no silu on node_dec2
    atomicAdd(&hsum[(size_t)(nmi / NPG) * HID + n], s);
  }
}

// ---- graph_dec: pred = w2 . silu(W1 @ hsum + b1) + b2 ----
__global__ __launch_bounds__(256) void graph_dec_kernel(const float* __restrict__ hsum,
                                                        const _Float16* __restrict__ W1,
                                                        const float* __restrict__ b1,
                                                        const float* __restrict__ w2,
                                                        const float* __restrict__ b2,
                                                        float* __restrict__ pred) {
  __shared__ _Float16 ldsA[16 * STRIDE_H];
  __shared__ float ldsT[16 * HID];
  __shared__ float parts[256];
  int tid = threadIdx.x, lane = tid & 31, w = tid >> 5;
  int t = blockIdx.x;
  for (int idx = tid; idx < 16 * HID; idx += 256) {
    int i = idx >> 7, k = idx & 127;
    ldsA[i * STRIDE_H + k] = (_Float16)hsum[(size_t)(t*16 + i) * HID + k];
  }
  v16h B1[4];
  #pragma unroll
  for (int kc = 0; kc < 4; ++kc) B1[kc] = load_b_packed(W1, kc*8 + w, lane);
  __syncthreads();

  int nloc = lane & 15, g = lane >> 4, n = w*16 + nloc, m = nloc;
  float bv = b1[n];
  v8f acc = {bv, bv, bv, bv, bv, bv, bv, bv};
  #pragma unroll
  for (int kc = 0; kc < 4; ++kc)
    acc = wmma16(load_a_lds(ldsA, STRIDE_H, m, g, kc*32), B1[kc], acc);
  #pragma unroll
  for (int r = 0; r < 8; ++r)
    ldsT[(r + g*8) * HID + n] = silu_f(acc[r]);
  __syncthreads();

  int i = tid >> 4, j = tid & 15;
  float p = 0.0f;
  #pragma unroll
  for (int kk = 0; kk < 8; ++kk) {
    int k = j*8 + kk;
    p += ldsT[i * HID + k] * w2[k];
  }
  parts[tid] = p;
  __syncthreads();
  if (tid < 16) {
    float s = b2[0];
    #pragma unroll
    for (int jj = 0; jj < 16; ++jj) s += parts[tid*16 + jj];
    pred[t*16 + tid] = s;
  }
}

extern "C" void kernel_launch(void* const* d_in, const int* in_sizes, int n_in,
                              void* d_out, int out_size, void* d_ws, size_t ws_size,
                              hipStream_t stream) {
  (void)in_sizes; (void)n_in; (void)out_size; (void)ws_size;
  // Inputs per setup_inputs() dict order; params flattened by sorted keys
  // (embedding, gcl[0..3]{edge1,edge2,node1,node2}, graph_dec1, graph_dec2,
  //  node_dec1, node_dec2), each leaf {b, w}.
  const float*     h0        = (const float*)d_in[0];
  const float*     xcoord    = (const float*)d_in[1];
  const long long* edges     = (const long long*)d_in[2];
  const float*     edge_attr = (const float*)d_in[3];
  const float*     node_mask = (const float*)d_in[4];
  const float*     edge_mask = (const float*)d_in[5];
  // d_in[6] = n_nodes scalar (compile-time NPG)
  int p = 7;
  const float* emb_b = (const float*)d_in[p++];
  const float* emb_w = (const float*)d_in[p++];
  const float *e1b[4], *e1w[4], *e2b[4], *e2w[4], *n1b[4], *n1w[4], *n2b[4], *n2w[4];
  for (int l = 0; l < 4; ++l) {
    e1b[l] = (const float*)d_in[p++]; e1w[l] = (const float*)d_in[p++];
    e2b[l] = (const float*)d_in[p++]; e2w[l] = (const float*)d_in[p++];
    n1b[l] = (const float*)d_in[p++]; n1w[l] = (const float*)d_in[p++];
    n2b[l] = (const float*)d_in[p++]; n2w[l] = (const float*)d_in[p++];
  }
  const float* gd1b = (const float*)d_in[p++];
  const float* gd1w = (const float*)d_in[p++];
  const float* gd2b = (const float*)d_in[p++];
  const float* gd2w = (const float*)d_in[p++];
  const float* nd1b = (const float*)d_in[p++];
  const float* nd1w = (const float*)d_in[p++];
  const float* nd2b = (const float*)d_in[p++];
  const float* nd2w = (const float*)d_in[p++];
  float* pred = (float*)d_out;

  // ---- workspace carve ----
  char* wsp = (char*)d_ws;
  auto take = [&](size_t bytes) { char* q = wsp; wsp += (bytes + 255) & ~(size_t)255; return q; };
  float*    hF     = (float*)take((size_t)NNODE * HID * 4);
  _Float16* hB     = (_Float16*)take((size_t)NNODE * HID * 2);
  float*    agg    = (float*)take((size_t)NNODE * HID * 4);
  float*    radial = (float*)take((size_t)NEDGE * 4);
  float*    hsum   = (float*)take((size_t)NBATCH * HID * 4);
  int*      e32    = (int*)take((size_t)2 * NEDGE * 4);
  _Float16* pEmb   = (_Float16*)take((size_t)32 * HID * 2);
  _Float16 *pE1[4], *pE2[4], *pN1[4], *pN2[4];
  for (int l = 0; l < 4; ++l) {
    pE1[l] = (_Float16*)take((size_t)K_EDGE_PAD * HID * 2);
    pE2[l] = (_Float16*)take((size_t)HID * HID * 2);
    pN1[l] = (_Float16*)take((size_t)K_NODE_PAD * HID * 2);
    pN2[l] = (_Float16*)take((size_t)HID * HID * 2);
  }
  _Float16* pD1 = (_Float16*)take((size_t)HID * HID * 2);
  _Float16* pD2 = (_Float16*)take((size_t)HID * HID * 2);
  _Float16* pG1 = (_Float16*)take((size_t)HID * HID * 2);

  int* rows = e32;
  int* cols = e32 + NEDGE;

  conv_edges<<<(2*NEDGE + 255)/256, 256, 0, stream>>>(edges, e32, 2*NEDGE);
  radial_kernel<<<(NEDGE + 255)/256, 256, 0, stream>>>(xcoord, rows, cols, radial);

  pack_weight<<<(32*HID)/256, 256, 0, stream>>>(emb_w, pEmb, 15, 32);
  for (int l = 0; l < 4; ++l) {
    pack_weight<<<(K_EDGE_PAD*HID)/256, 256, 0, stream>>>(e1w[l], pE1[l], K_EDGE, K_EDGE_PAD);
    pack_weight<<<(HID*HID)/256,        256, 0, stream>>>(e2w[l], pE2[l], HID, HID);
    pack_weight<<<(K_NODE_PAD*HID)/256, 256, 0, stream>>>(n1w[l], pN1[l], K_NODE, K_NODE_PAD);
    pack_weight<<<(HID*HID)/256,        256, 0, stream>>>(n2w[l], pN2[l], HID, HID);
  }
  pack_weight<<<(HID*HID)/256, 256, 0, stream>>>(nd1w, pD1, HID, HID);
  pack_weight<<<(HID*HID)/256, 256, 0, stream>>>(nd2w, pD2, HID, HID);
  pack_weight<<<(HID*HID)/256, 256, 0, stream>>>(gd1w, pG1, HID, HID);

  embed_kernel<<<NODE_TILES, 256, 0, stream>>>(h0, pEmb, emb_b, hF, hB);

  for (int l = 0; l < 4; ++l) {
    hipMemsetAsync(agg, 0, (size_t)NNODE * HID * 4, stream);
    edge_kernel<<<EDGE_GRID, 256, 0, stream>>>(hB, radial, edge_attr, edge_mask, rows, cols,
                                               pE1[l], e1b[l], pE2[l], e2b[l], agg);
    node_kernel<<<NODE_GRID, 256, 0, stream>>>(h0, agg, pN1[l], n1b[l], pN2[l], n2b[l], hF, hB);
  }

  hipMemsetAsync(hsum, 0, (size_t)NBATCH * HID * 4, stream);
  node_dec_kernel<<<NODE_TILES, 256, 0, stream>>>(hB, node_mask, pD1, nd1b, pD2, nd2b, hsum);
  graph_dec_kernel<<<NBATCH/16, 256, 0, stream>>>(hsum, pG1, gd1b, gd2w, gd2b, pred);
}
